// MultiHeadAttention_76854144794983
// MI455X (gfx1250) — matrix operand order repaired and checked by reading, not compiled, not run on hardware
//
#include <hip/hip_runtime.h>
#include <hip/hip_bf16.h>
#include <stdint.h>

// ---------------------------------------------------------------------------
// MI455X (gfx1250) fused multi-head attention, bf16 WMMA pipeline.
//   x[4,2048,1024] @ w_qkv[1024,3072] -> qkv
//   per (b,h): flash attention over 2048 keys, dh=64
//   attn_out[8192,1024] @ w_out[1024,1024] + b_out -> out (fp32)
// Matrix math: v_wmma_f32_16x16x32_bf16 (fp32 accumulate).
// Data movement: GLOBAL_LOAD_ASYNC_TO_LDS_B128 double-buffered staging
// (ASYNCcnt) + register prefetch for tiles that need an LDS transpose.
// ---------------------------------------------------------------------------

typedef __bf16 bf16;
typedef __attribute__((ext_vector_type(16))) __bf16        v16bf;
typedef __attribute__((ext_vector_type(4)))  __bf16        v4bf;
typedef __attribute__((ext_vector_type(8)))  float         v8f;
typedef __attribute__((ext_vector_type(4)))  float         f32x4;
typedef __attribute__((ext_vector_type(4)))  unsigned int  u32x4;

union FragU { v16bf v; u32x4 q[2]; };
union VecU  { u32x4 q; bf16 h[8]; };

// ---- CDNA5 async global->LDS DMA (ASYNCcnt) -------------------------------
// Shared-aperture flat pointers keep the LDS byte offset in the low dword,
// so truncating a generic __shared__ pointer yields the DS address.
static __device__ __forceinline__ unsigned lds_off(const void* p) {
  return (unsigned)(uintptr_t)p;
}
static __device__ __forceinline__ void async_copy_b128(unsigned lds_byte_off,
                                                       const void* gptr) {
  asm volatile("global_load_async_to_lds_b128 %0, %1, off"
               :: "v"(lds_byte_off), "v"((unsigned long long)(uintptr_t)gptr)
               : "memory");
}
template <int N>
static __device__ __forceinline__ void wait_asynccnt() {
  asm volatile("s_wait_asynccnt %0" :: "i"(N) : "memory");
}

// A-matrix 16-bit fragment (16x32): lane holds K = half*16+0..15 contiguous.
// p points at K = half*16.
static __device__ __forceinline__ v16bf load_frag_a(const bf16* p) {
  FragU f;
  f.q[0] = *(const u32x4*)(p);
  f.q[1] = *(const u32x4*)(p + 8);
  return f.v;
}
// B-matrix 16-bit fragment (32x16): lane holds K = half*16+0..15 contiguous.
static __device__ __forceinline__ v16bf load_frag_b(const bf16* p) {
  FragU f;
  f.q[0] = *(const u32x4*)(p);
  f.q[1] = *(const u32x4*)(p + 8);
  return f.v;
}
static __device__ __forceinline__ v8f wmma_bf16(v16bf a, v16bf b, v8f c) {
  return __builtin_amdgcn_wmma_f32_16x16x32_bf16(
      false, a, false, b, (short)0, c, false, false);
}

// ---------------------------------------------------------------------------
// fp32 -> bf16 conversion (vectorized, grid-stride; n multiple of 4)
// ---------------------------------------------------------------------------
__global__ void cvt_f32_bf16_kernel(const float* __restrict__ in,
                                    bf16* __restrict__ out, long long n4) {
  long long i = (long long)blockIdx.x * blockDim.x + threadIdx.x;
  const long long stride = (long long)gridDim.x * blockDim.x;
  for (; i < n4; i += stride) {
    const f32x4 v = *(const f32x4*)(in + i * 4);
    v4bf o;
    o.x = (bf16)v.x; o.y = (bf16)v.y; o.z = (bf16)v.z; o.w = (bf16)v.w;
    *(v4bf*)(out + i * 4) = o;
  }
}

// ---------------------------------------------------------------------------
// 128x128 block GEMM, BK=32, 8 waves (2x4), wave tile 64x32 (4x2 WMMA tiles).
// A tiles: async-DMA double buffered.  B tiles: register prefetch + LDS
// transpose scatter (async DMA cannot transpose).
// ---------------------------------------------------------------------------
__global__ __launch_bounds__(256) void gemm_bf16_128x128(
    const bf16* __restrict__ A, const bf16* __restrict__ B,
    int K, int N, bf16* __restrict__ Cbf, float* __restrict__ Cf,
    const float* __restrict__ bias) {
  constexpr int STR = 40;  // 80B rows: 16B aligned fragments, bank-spread
  __shared__ bf16 lsa[2][128 * STR];   // [m][k], ping-pong, async-filled
  __shared__ bf16 lsb[128 * STR];      // [n][k] (transposed from global)

  const int tid = threadIdx.x;
  const int wid = tid >> 5, lane = tid & 31;
  const int lh = lane >> 4, lm = lane & 15;
  const int wm = wid >> 2, wn = wid & 3;
  const int m0 = blockIdx.y * 128, n0 = blockIdx.x * 128;

  // Per-thread staging slots (2 x 16B chunks each for A and B).
  const int arow = tid >> 2, akc = (tid & 3) * 8;   // A rows arow, arow+64
  const int bk0 = tid >> 4, bnc = (tid & 15) * 8;   // B k rows bk0, bk0+16

  v8f acc[4][2] = {};

  // Prologue: async A tile0 into lsa[0]; B tile0 into registers.
  #pragma unroll
  for (int u = 0; u < 2; ++u) {
    const int row = arow + u * 64;
    async_copy_b128(lds_off(&lsa[0][row * STR + akc]),
                    A + (size_t)(m0 + row) * K + akc);
  }
  VecU breg[2];
  #pragma unroll
  for (int u = 0; u < 2; ++u)
    breg[u].q = *(const u32x4*)(B + (size_t)(bk0 + u * 16) * N + n0 + bnc);

  const int T = K / 32;
  for (int t = 0; t < T; ++t) {
    const int cb = t & 1;
    const bool has_next = (t + 1 < T);
    __syncthreads();  // all reads of lsa[1-cb] and lsb from iter t-1 done

    // Prefetch tile t+1: async A -> lsa[1-cb]; B -> registers.
    VecU bnext[2];
    if (has_next) {
      const int kn = (t + 1) * 32;
      #pragma unroll
      for (int u = 0; u < 2; ++u) {
        const int row = arow + u * 64;
        async_copy_b128(lds_off(&lsa[1 - cb][row * STR + akc]),
                        A + (size_t)(m0 + row) * K + kn + akc);
      }
      #pragma unroll
      for (int u = 0; u < 2; ++u)
        bnext[u].q = *(const u32x4*)(B + (size_t)(kn + bk0 + u * 16) * N + n0 + bnc);
    }

    // Scatter current B registers into lsb as [n][k].
    #pragma unroll
    for (int u = 0; u < 2; ++u) {
      const int kk = bk0 + u * 16;
      #pragma unroll
      for (int j = 0; j < 8; ++j) lsb[(bnc + j) * STR + kk] = breg[u].h[j];
    }

    // Current A async ops (issued last iter) must be complete; the 2 just
    // issued for the next tile stay in flight across the WMMA block.
    if (has_next) wait_asynccnt<2>(); else wait_asynccnt<0>();
    __syncthreads();

    v16bf afr[4], bfr[2];
    #pragma unroll
    for (int i = 0; i < 4; ++i)
      afr[i] = load_frag_a(&lsa[cb][(wm * 64 + i * 16 + lm) * STR + lh * 16]);
    #pragma unroll
    for (int j = 0; j < 2; ++j)
      bfr[j] = load_frag_b(&lsb[(wn * 32 + j * 16 + lm) * STR + lh * 16]);

    #pragma unroll
    for (int i = 0; i < 4; ++i)
      #pragma unroll
      for (int j = 0; j < 2; ++j)
        acc[i][j] = wmma_bf16(afr[i], bfr[j], acc[i][j]);

    if (has_next) { breg[0] = bnext[0]; breg[1] = bnext[1]; }
  }

  // Epilogue. C/D layout: VGPR r -> row r + 8*half, lane&15 -> col.
  #pragma unroll
  for (int i = 0; i < 4; ++i) {
    const int row0 = m0 + wm * 64 + i * 16 + lh * 8;
    #pragma unroll
    for (int j = 0; j < 2; ++j) {
      const int col = n0 + wn * 32 + j * 16 + lm;
      const float bv = (Cf && bias) ? bias[col] : 0.0f;
      #pragma unroll
      for (int r = 0; r < 8; ++r) {
        const size_t idx = (size_t)(row0 + r) * N + col;
        const float v = acc[i][j][r] + bv;
        if (Cf) Cf[idx] = v;
        else    Cbf[idx] = (bf16)v;
      }
    }
  }
}

// ---------------------------------------------------------------------------
// Fused flash attention.  Block = (q-tile of 128, head, batch); 8 waves,
// each wave owns 16 query rows; dh = 64; key tiles of 32.
// qkv rows are [q(1024) | k(1024) | v(1024)] bf16, row stride 3072.
// V tiles are register-prefetched one iteration ahead.
// ---------------------------------------------------------------------------
__global__ __launch_bounds__(256) void attn_fused_kernel(
    const bf16* __restrict__ qkv, bf16* __restrict__ attn_out) {
  constexpr int STR = 40;
  __shared__ bf16 lds_v[64 * STR];        // V tile transposed: [dh][key]
  __shared__ bf16 lds_p[8][16 * STR];     // per-wave P tile: [m][key]

  const int tid = threadIdx.x;
  const int wid = tid >> 5, lane = tid & 31;
  const int lh = lane >> 4, lm = lane & 15;
  const int qt = blockIdx.x, h = blockIdx.y, b = blockIdx.z;
  const bf16* base = qkv + (size_t)b * 2048 * 3072;
  const int q0 = qt * 128 + wid * 16;
  const float scale = 0.125f;  // 1/sqrt(64)

  // Q fragments (16 rows x 64 dh -> two 16x32 A-frags) live in registers.
  v16bf qfr[2];
  #pragma unroll
  for (int s = 0; s < 2; ++s)
    qfr[s] = load_frag_a(base + (size_t)(q0 + lm) * 3072 + h * 64 + s * 32 + lh * 16);

  float m_run[8], l_run[8];
  #pragma unroll
  for (int r = 0; r < 8; ++r) { m_run[r] = -1e30f; l_run[r] = 0.0f; }
  v8f oacc[4] = {};

  // V staging slot for this thread: key row (tid>>3), dh chunk (tid&7)*8.
  const int vkey = tid >> 3, vdc = (tid & 7) * 8;
  VecU vreg;
  vreg.q = *(const u32x4*)(base + (size_t)vkey * 3072 + 2048 + h * 64 + vdc);

  for (int kv = 0; kv < 2048; kv += 32) {
    __syncthreads();  // previous iteration's reads of lds_v done

    // Prefetch next V tile into registers (overlaps this tile's compute).
    VecU vnext;
    const bool has_next = (kv + 32 < 2048);
    if (has_next)
      vnext.q = *(const u32x4*)(base + (size_t)(kv + 32 + vkey) * 3072 +
                                2048 + h * 64 + vdc);

    // Scatter current V tile transposed to [dh][key].
    #pragma unroll
    for (int j = 0; j < 8; ++j) lds_v[(vdc + j) * STR + vkey] = vreg.h[j];
    __syncthreads();

    // S = Q @ K^T.  K rows [key][dh] are already WMMA-B layout.
    v8f sacc[2] = {};
    #pragma unroll
    for (int t2 = 0; t2 < 2; ++t2) {
      const bf16* krow = base + (size_t)(kv + t2 * 16 + lm) * 3072 + 1024 + h * 64;
      #pragma unroll
      for (int s = 0; s < 2; ++s) {
        v16bf kf = load_frag_b(krow + s * 32 + lh * 16);
        sacc[t2] = wmma_bf16(qfr[s], kf, sacc[t2]);
      }
    }

    // Online softmax.  Row r+8*lh lives in VGPR r across 16 lanes of a half.
    float p0[8], p1[8];
    #pragma unroll
    for (int r = 0; r < 8; ++r) {
      const float a0 = sacc[0][r] * scale;
      const float a1 = sacc[1][r] * scale;
      float mx = fmaxf(a0, a1);
      #pragma unroll
      for (int d = 1; d < 16; d <<= 1) mx = fmaxf(mx, __shfl_xor(mx, d, 32));
      const float mnew = fmaxf(m_run[r], mx);
      const float corr = __expf(m_run[r] - mnew);
      const float e0 = __expf(a0 - mnew);
      const float e1 = __expf(a1 - mnew);
      float sum = e0 + e1;
      #pragma unroll
      for (int d = 1; d < 16; d <<= 1) sum += __shfl_xor(sum, d, 32);
      l_run[r] = l_run[r] * corr + sum;
      m_run[r] = mnew;
      p0[r] = e0; p1[r] = e1;
      #pragma unroll
      for (int t = 0; t < 4; ++t) oacc[t][r] = oacc[t][r] * corr;
    }

    // C-layout -> A-layout for P via per-wave LDS round trip.
    bf16* pw = &lds_p[wid][0];
    #pragma unroll
    for (int r = 0; r < 8; ++r) {
      const int row = r + 8 * lh;
      pw[row * STR + lm]      = (bf16)p0[r];
      pw[row * STR + 16 + lm] = (bf16)p1[r];
    }
    v16bf pf = load_frag_a(pw + lm * STR + lh * 16);

    // O += P @ V  (V frags from transposed LDS tile).
    #pragma unroll
    for (int t = 0; t < 4; ++t) {
      v16bf vf = load_frag_b(&lds_v[(t * 16 + lm) * STR + lh * 16]);
      oacc[t] = wmma_bf16(pf, vf, oacc[t]);
    }

    if (has_next) vreg = vnext;
  }

  // Normalize and store bf16 [b*2048+row][h*64+col].
  float inv[8];
  #pragma unroll
  for (int r = 0; r < 8; ++r) inv[r] = 1.0f / l_run[r];
  #pragma unroll
  for (int t = 0; t < 4; ++t) {
    const int col = h * 64 + t * 16 + lm;
    #pragma unroll
    for (int r = 0; r < 8; ++r) {
      const int row = q0 + r + 8 * lh;
      attn_out[((size_t)b * 2048 + row) * 1024 + col] = (bf16)(oacc[t][r] * inv[r]);
    }
  }
}

// ---------------------------------------------------------------------------
// Launch
// ---------------------------------------------------------------------------
extern "C" void kernel_launch(void* const* d_in, const int* in_sizes, int n_in,
                              void* d_out, int out_size, void* d_ws, size_t ws_size,
                              hipStream_t stream) {
  (void)in_sizes; (void)n_in; (void)out_size; (void)ws_size;
  const float* x     = (const float*)d_in[0];   // [4,2048,1024]
  const float* w_qkv = (const float*)d_in[1];   // [1024,3072]
  const float* w_out = (const float*)d_in[2];   // [1024,1024]
  const float* b_out = (const float*)d_in[3];   // [1024]
  float* out = (float*)d_out;                   // [4,2048,1024] fp32

  // Workspace layout (bytes): ~92 MB total.
  char* ws = (char*)d_ws;
  bf16* x_bf    = (bf16*)(ws + 0);               // 8192*1024*2  = 16 MiB
  bf16* wq_bf   = (bf16*)(ws + 16777216);        // 1024*3072*2  = 6 MiB
  bf16* wo_bf   = (bf16*)(ws + 23068672);        // 1024*1024*2  = 2 MiB
  bf16* qkv_bf  = (bf16*)(ws + 25165824);        // 8192*3072*2  = 48 MiB
  bf16* attn_bf = (bf16*)(ws + 75497472);        // 8192*1024*2  = 16 MiB

  cvt_f32_bf16_kernel<<<1024, 256, 0, stream>>>(x, x_bf, (long long)8192 * 1024 / 4);
  cvt_f32_bf16_kernel<<<512, 256, 0, stream>>>(w_qkv, wq_bf, (long long)1024 * 3072 / 4);
  cvt_f32_bf16_kernel<<<256, 256, 0, stream>>>(w_out, wo_bf, (long long)1024 * 1024 / 4);

  // qkv = x @ w_qkv   (M=8192, N=3072, K=1024)
  gemm_bf16_128x128<<<dim3(24, 64), 256, 0, stream>>>(
      x_bf, wq_bf, 1024, 3072, qkv_bf, nullptr, nullptr);

  // fused attention per (q-tile, head, batch)
  attn_fused_kernel<<<dim3(16, 16, 4), 256, 0, stream>>>(qkv_bf, attn_bf);

  // out = attn @ w_out + b_out  (M=8192, N=1024, K=1024), fp32 output
  gemm_bf16_128x128<<<dim3(8, 64), 256, 0, stream>>>(
      attn_bf, wo_bf, 1024, 1024, nullptr, out, b_out);
}
